// MultiHeadAttention_11811160064231
// MI455X (gfx1250) — matrix operand order repaired and checked by reading, not compiled, not run on hardware
//
#include <hip/hip_runtime.h>
#include <hip/hip_bf16.h>

typedef __attribute__((ext_vector_type(16))) _Float16 v16h;
typedef __attribute__((ext_vector_type(8)))  _Float16 v8h;
typedef __attribute__((ext_vector_type(8)))  float    v8f;

#define S_LEN   2048
#define D_MODEL 1024
#define NHEADS  16
#define DK      64
#define BATCH   4
#define M_ROWS  (BATCH * S_LEN)   // 8192

__device__ __forceinline__ int lane_id() { return (int)(threadIdx.x & 31u); }

__device__ __forceinline__ v8f wmma_f16(v16h a, v16h b, v8f c) {
  return __builtin_amdgcn_wmma_f32_16x16x32_f16(false, a, false, b, (short)0, c,
                                                false, false);
}

// CDNA5 async global->LDS copy (GV mode): VDST = LDS byte address (flat addr
// low 32 bits == LDS offset per ISA 10.2 aperture map), VADDR = 64-bit global.
__device__ __forceinline__ void async_copy16_to_lds(const void* gsrc, void* ldst) {
  const uint64_t gaddr = (uint64_t)(uintptr_t)gsrc;
  const uint32_t laddr = (uint32_t)(uintptr_t)ldst;
  asm volatile("global_load_async_to_lds_b128 %0, %1, off"
               :: "v"(laddr), "v"(gaddr)
               : "memory");
}

// ---- Fragment loaders per CDNA5 ISA 7.12.2 VGPR layouts (wave32) ----

// A fragment 16x32 from row-major src; j-groups of 8 are contiguous -> 16B loads.
template <typename T>
__device__ __forceinline__ v16h load_a_frag(const T* __restrict__ A, int lda) {
  const int lane = lane_id();
  const int m  = lane & 15;
  const int kb = (lane >> 4) << 3;          // 0 or 8
  v16h a;
#pragma unroll
  for (int j = 0; j < 16; ++j) {
    const int k = ((j >> 3) << 4) + kb + (j & 7);
    a[j] = (_Float16)A[(size_t)m * lda + k];
  }
  return a;
}

// B fragment 32x16 where logical B[k][n] = src[n*ld + k] (K-major src);
// each lane reads 2x contiguous 16B -> global_load_b128 / ds_load_b128.
template <typename T>
__device__ __forceinline__ v16h load_bt_frag(const T* __restrict__ Bt, int ld) {
  const int lane = lane_id();
  const int n  = lane & 15;
  const int kb = (lane >> 4) << 3;          // 0 or 8
  v16h b;
#pragma unroll
  for (int j = 0; j < 16; ++j) b[j] = (_Float16)Bt[(size_t)n * ld + ((j >> 3) << 4) + kb + (j & 7)];
  return b;
}

// ---------------- Weight transpose+convert: Wt[n][k] = (f16)W[k][n] ----------------
__global__ __launch_bounds__(256) void wtrans_kernel(const float* __restrict__ W,
                                                     _Float16* __restrict__ Wt) {
  const int idx = blockIdx.x * 256 + (int)threadIdx.x;   // 1M threads
  const int k = idx >> 10;
  const int n = idx & 1023;                               // consecutive: coalesced read
  Wt[(size_t)n * D_MODEL + k] = (_Float16)W[(size_t)k * D_MODEL + n];
}

// ---------------- GEMM: C[8192x1024] = A[8192x1024] * Wt^T, wave = 16x64 strip ------
template <bool A_F32, bool SCATTER>
__global__ __launch_bounds__(256) void gemm_kernel(const void* __restrict__ Ap,
                                                   const _Float16* __restrict__ Wt,
                                                   void* __restrict__ Outp) {
  const int w  = blockIdx.x * 8 + (int)(threadIdx.x >> 5);  // 8192 waves total
  const int m0 = (w >> 4) << 4;                             // 512 row tiles
  const int n0 = (w & 15) << 6;                             // 16 col strips
  const float*    Af = (const float*)Ap;
  const _Float16* Ah = (const _Float16*)Ap;

  v8f c[4] = {};
  for (int k0 = 0; k0 < D_MODEL; k0 += 32) {
    v16h a;
    if constexpr (A_F32) a = load_a_frag(Af + (size_t)m0 * D_MODEL + k0, D_MODEL);
    else                 a = load_a_frag(Ah + (size_t)m0 * D_MODEL + k0, D_MODEL);
#pragma unroll
    for (int t = 0; t < 4; ++t) {
      v16h b = load_bt_frag(Wt + (size_t)(n0 + t * 16) * D_MODEL + k0, D_MODEL);
      c[t] = wmma_f16(a, b, c[t]);
    }
  }

  const int lane  = lane_id();
  const int ncol  = lane & 15;
  const int rbase = (lane >> 4) << 3;
#pragma unroll
  for (int t = 0; t < 4; ++t) {
#pragma unroll
    for (int r = 0; r < 8; ++r) {
      const int m = m0 + rbase + r;
      const int n = n0 + t * 16 + ncol;
      if constexpr (SCATTER) {
        // raw reshape [B,S,H*dk]->[B,H,S,dk]: row=h*128+s_hi, col=s_lo*64+d
        _Float16* heads = (_Float16*)Outp;
        const int b   = m >> 11;
        const int rb  = m & 2047;
        const int h   = rb >> 7;
        const int shi = rb & 127;
        const int slo = n >> 6;
        const int d   = n & 63;
        const int s   = shi * 16 + slo;
        heads[((size_t)((b * NHEADS + h) * S_LEN + s)) * DK + d] = (_Float16)c[t][r];
      } else {
        ((float*)Outp)[(size_t)m * D_MODEL + n] = c[t][r];
      }
    }
  }
}

// ---------------- Flash attention ----------------
// Block = 8 waves = 8 consecutive 16-row q tiles of ONE (b,h); the 32-key K/V
// block is staged in LDS once per iteration and shared by all 8 waves.
#define VT_LD 56   // 112B rows: 16B-aligned, bank-conflict-free for 16-lane column reads

__global__ __launch_bounds__(256) void attn_kernel(const _Float16* __restrict__ Qh,
                                                   const _Float16* __restrict__ Kh,
                                                   const _Float16* __restrict__ Vh,
                                                   _Float16* __restrict__ Ctx) {
  __shared__ _Float16 kt[32][64];          // K block, row-major (async target), 4KB
  __shared__ _Float16 vt[64][VT_LD];       // V block, transposed [d][key], 7KB
  __shared__ _Float16 pbuf[8][16][40];     // per-wave P tile, 10KB

  const int wv   = (int)(threadIdx.x >> 5);
  const int tid  = (int)threadIdx.x;
  const int bh   = blockIdx.x >> 4;                  // 64 (b,h) pairs
  const int qt   = (blockIdx.x & 15) * 8 + wv;       // 128 q tiles per (b,h)
  const int lane = lane_id();
  const int ncol  = lane & 15;
  const int rbase = (lane >> 4) << 3;

  const size_t base = (size_t)bh * S_LEN * DK;
  const _Float16* q = Qh + base + (size_t)(qt * 16) * DK;
  const _Float16* k = Kh + base;
  const _Float16* v = Vh + base;

  // Q fragments over d=0..31 / 32..63, pre-scaled by 1/sqrt(dk)=0.125 (exact pow2)
  v16h aq0 = load_a_frag(q, DK);
  v16h aq1 = load_a_frag(q + 32, DK);
#pragma unroll
  for (int j = 0; j < 16; ++j) { aq0[j] *= (_Float16)0.125f; aq1[j] *= (_Float16)0.125f; }

  float m_run[8], l_run[8];
#pragma unroll
  for (int r = 0; r < 8; ++r) { m_run[r] = -1e30f; l_run[r] = 0.0f; }
  v8f o[4] = {};

  // Per-thread staging assignments: 256 threads x 16B cover one 32x64 f16 tile.
  const int krow = tid >> 3, koff = (tid & 7) * 8;   // K: direct async copy
  const int vrow = tid >> 3, voff = (tid & 7) * 8;   // V: transpose via registers

  for (int kb = 0; kb < S_LEN; kb += 32) {
    __syncthreads();   // previous iteration's LDS reads complete before refill

    // ---- K block -> LDS via CDNA5 async DMA (ASYNCcnt-tracked) ----
    async_copy16_to_lds(k + (size_t)(kb + krow) * DK + koff, &kt[krow][koff]);

    // ---- V block -> LDS transposed [d][key] so frags are contiguous ----
    {
      const v8h tmp = *(const v8h*)(v + (size_t)(kb + vrow) * DK + voff);
#pragma unroll
      for (int i = 0; i < 8; ++i) vt[voff + i][vrow] = tmp[i];
    }
    asm volatile("s_wait_asynccnt 0x0" ::: "memory");
    asm volatile("s_wait_dscnt 0x0" ::: "memory");
    __syncthreads();

    // ---- scores S = (q/8) @ K^T : two 16x16 frags over the 32-key block ----
    v8f s0 = {}, s1 = {};
    {
      v16h b00 = load_bt_frag(&kt[0][0],   64);
      v16h b01 = load_bt_frag(&kt[0][32],  64);
      s0 = wmma_f16(aq0, b00, s0);  s0 = wmma_f16(aq1, b01, s0);
      v16h b10 = load_bt_frag(&kt[16][0],  64);
      v16h b11 = load_bt_frag(&kt[16][32], 64);
      s1 = wmma_f16(aq0, b10, s1);  s1 = wmma_f16(aq1, b11, s1);
    }

    // ---- online softmax; rows live in 16-lane halves -> xor-shuffle reductions ----
    float alpha_s[8];
#pragma unroll
    for (int r = 0; r < 8; ++r) {
      float vmax = fmaxf(s0[r], s1[r]);
#pragma unroll
      for (int off = 1; off < 16; off <<= 1) vmax = fmaxf(vmax, __shfl_xor(vmax, off, 32));
      const float mnew  = fmaxf(m_run[r], vmax);
      const float alpha = __expf(m_run[r] - mnew);
      m_run[r] = mnew;
      const float p0 = __expf(s0[r] - mnew);
      const float p1 = __expf(s1[r] - mnew);
      float rs = p0 + p1;
#pragma unroll
      for (int off = 1; off < 16; off <<= 1) rs += __shfl_xor(rs, off, 32);
      l_run[r]  = l_run[r] * alpha + rs;
      alpha_s[r] = alpha;
      pbuf[wv][rbase + r][ncol]      = (_Float16)p0;
      pbuf[wv][rbase + r][16 + ncol] = (_Float16)p1;
    }
#pragma unroll
    for (int t = 0; t < 4; ++t)
#pragma unroll
      for (int r = 0; r < 8; ++r) o[t][r] *= alpha_s[r];

    // wave-local C-layout -> A-layout via per-wave LDS tile
    asm volatile("s_wait_dscnt 0x0" ::: "memory");
    v16h ap = load_a_frag(&pbuf[wv][0][0], 40);

    // ---- O += P(16x32) @ V(32x64), V frags from transposed LDS tile ----
#pragma unroll
    for (int t = 0; t < 4; ++t) {
      v16h bv = load_bt_frag(&vt[t * 16][0], VT_LD);
      o[t] = wmma_f16(ap, bv, o[t]);
    }
  }

  // ---- normalize; store ctx as [B, S, H*dk] (standard transpose back) ----
  const int b = bh >> 4, h = bh & 15;
#pragma unroll
  for (int r = 0; r < 8; ++r) {
    const float inv = 1.0f / l_run[r];
    const int   s   = qt * 16 + rbase + r;
#pragma unroll
    for (int t = 0; t < 4; ++t) {
      Ctx[((size_t)(b * S_LEN + s)) * D_MODEL + h * DK + t * 16 + ncol] =
          (_Float16)(o[t][r] * inv);
    }
  }
}

extern "C" void kernel_launch(void* const* d_in, const int* in_sizes, int n_in,
                              void* d_out, int out_size, void* d_ws, size_t ws_size,
                              hipStream_t stream) {
  (void)in_sizes; (void)n_in; (void)out_size; (void)ws_size;
  const float* Q   = (const float*)d_in[0];
  const float* K   = (const float*)d_in[1];
  const float* V   = (const float*)d_in[2];
  const float* WQ  = (const float*)d_in[3];
  const float* WK  = (const float*)d_in[4];
  const float* WV  = (const float*)d_in[5];
  const float* Wfc = (const float*)d_in[6];
  float* out = (float*)d_out;

  const size_t elems = (size_t)M_ROWS * D_MODEL;       // 8M
  const size_t welems = (size_t)D_MODEL * D_MODEL;     // 1M
  _Float16* q_heads = (_Float16*)d_ws;                 // 16MB  [B,H,S,dk]
  _Float16* k_heads = q_heads + elems;                 // 16MB
  _Float16* v_heads = k_heads + elems;                 // 16MB
  _Float16* ctxh    = v_heads + elems;                 // 16MB  [B,S,D]
  _Float16* WQt     = ctxh + elems;                    // 2MB each, f16 [n][k]
  _Float16* WKt     = WQt + welems;
  _Float16* WVt     = WKt + welems;
  _Float16* Wfct    = WVt + welems;

  dim3 blk(256);
  wtrans_kernel<<<dim3(4096), blk, 0, stream>>>(WQ,  WQt);
  wtrans_kernel<<<dim3(4096), blk, 0, stream>>>(WK,  WKt);
  wtrans_kernel<<<dim3(4096), blk, 0, stream>>>(WV,  WVt);
  wtrans_kernel<<<dim3(4096), blk, 0, stream>>>(Wfc, Wfct);

  dim3 grd(1024);                                      // 8 waves/WG, 8192 wave-tasks
  gemm_kernel<true,  true ><<<grd, blk, 0, stream>>>(Q, WQt, q_heads);
  gemm_kernel<true,  true ><<<grd, blk, 0, stream>>>(K, WKt, k_heads);
  gemm_kernel<true,  true ><<<grd, blk, 0, stream>>>(V, WVt, v_heads);
  attn_kernel<<<grd, blk, 0, stream>>>(q_heads, k_heads, v_heads, ctxh);
  gemm_kernel<false, false><<<grd, blk, 0, stream>>>(ctxh, Wfct, out);
}
